// LiquidMambaBlock_39771397161661
// MI455X (gfx1250) — matrix operand order repaired and checked by reading, not compiled, not run on hardware
//
#include <hip/hip_runtime.h>

// ---------------- problem constants (match reference) ----------------
#define D_MODEL   1024
#define D_INNER   2048
#define D_STATE   16
#define D_CONV    4
#define DT_RANK   64
#define BATCH     4
#define SEQ       2048
#define XDBL_W    (DT_RANK + 2 * D_STATE)   // 96
#define MROWS     (BATCH * SEQ)             // 8192

// ---------------- CDNA5 WMMA types ----------------
typedef __attribute__((ext_vector_type(8)))  __bf16 v8bf;
typedef __attribute__((ext_vector_type(16))) __bf16 v16bf;
typedef __attribute__((ext_vector_type(8)))  float  v8f;

struct Halves { v8bf lo; v8bf hi; };
union FragU { v16bf v; Halves p; };

__device__ __forceinline__ unsigned short f32_to_bf16(float f) {
  unsigned int u = __float_as_uint(f);
  u += 0x7FFFu + ((u >> 16) & 1u);          // round-to-nearest-even
  return (unsigned short)(u >> 16);
}

// ---- gfx1250 async copy: global -> LDS, 16 bytes per lane (ASYNCcnt) ----
__device__ __forceinline__ void async_copy_b128(unsigned int lds_byte_addr,
                                                const void* gaddr) {
  unsigned long long ga = (unsigned long long)(uintptr_t)gaddr;
  asm volatile("global_load_async_to_lds_b128 %0, %1, off"
               :: "v"(lds_byte_addr), "v"(ga) : "memory");
}

// ---------------- WMMA GEMM: C[M,N] = A[M,K(lda)] * Bt[N,K]^T ----------------
// A: bf16 row-major [M][lda]; Bt: bf16 pre-transposed [N][K]; C: fp32 [M][N].
// Double-buffered async LDS staging; 8 waves, wave tile 32x64, block tile 64x256.
#define BM 64
#define BN 256
#define BK 32
#define LDS_K 40      // padded row stride in ushorts (80B: 16B aligned, bank stride 20)
#define SA_SZ (BM * LDS_K)
#define SB_SZ (BN * LDS_K)

__global__ __launch_bounds__(256) void wmma_gemm_bf16_kernel(
    const unsigned short* __restrict__ A, const unsigned short* __restrict__ Bt,
    float* __restrict__ C, unsigned short* __restrict__ Cbf,
    int M, int N, int K, int lda)
{
  __shared__ __align__(16) unsigned short sA[2][SA_SZ];   // [m][k]
  __shared__ __align__(16) unsigned short sB[2][SB_SZ];   // [n][k]

  const int row0 = blockIdx.y * BM;
  const int col0 = blockIdx.x * BN;
  const int tid  = threadIdx.x;
  const int lane = tid & 31;
  const int wave = tid >> 5;
  const int wm   = wave >> 2;   // 0..1 : wave M sub-tile (32 rows)
  const int wn   = wave & 3;    // 0..3 : wave N sub-tile (64 cols)
  const int g    = lane >> 4;
  const int li   = lane & 15;

  // ---- per-thread staging addresses (constant across k-tiles) ----
  const int ar = tid >> 2;            // A row 0..63
  const int ac = tid & 3;             // A k-chunk 0..3
  const unsigned short* pA = A + (size_t)(row0 + ar) * lda + ac * 8;
  const unsigned int aoff = (unsigned)(ar * LDS_K + ac * 8) * 2;

  const unsigned short* pB[4];
  unsigned int boff[4];
#pragma unroll
  for (int j = 0; j < 4; ++j) {
    const int lin = tid + j * 256;
    const int n = lin >> 2, c = lin & 3;
    const int ncl = min(col0 + n, N - 1);   // clamp: no divergence, garbage confined
    pB[j] = Bt + (size_t)ncl * K + c * 8;
    boff[j] = (unsigned)(n * LDS_K + c * 8) * 2;
  }

  const unsigned int uA[2] = { (unsigned int)(uintptr_t)(void*)&sA[0][0],
                               (unsigned int)(uintptr_t)(void*)&sA[1][0] };
  const unsigned int uB[2] = { (unsigned int)(uintptr_t)(void*)&sB[0][0],
                               (unsigned int)(uintptr_t)(void*)&sB[1][0] };

  auto issue_tile = [&](int b, int k0) {
    async_copy_b128(uA[b] + aoff, pA + k0);
#pragma unroll
    for (int j = 0; j < 4; ++j)
      async_copy_b128(uB[b] + boff[j], pB[j] + k0);
  };

  v8f acc[2][4];
  {
    v8f z = {};
#pragma unroll
    for (int i = 0; i < 2; ++i)
#pragma unroll
      for (int j = 0; j < 4; ++j) acc[i][j] = z;
  }

  const int ktiles = K / BK;          // all K are multiples of 32 here
  issue_tile(0, 0);                   // prologue

  for (int kt = 0; kt < ktiles; ++kt) {
    const int cur = kt & 1;
    if (kt + 1 < ktiles) {
      issue_tile(cur ^ 1, (kt + 1) * BK);
      asm volatile("s_wait_asynccnt 0x5" ::: "memory");  // tile kt done, kt+1 in flight
    } else {
      asm volatile("s_wait_asynccnt 0x0" ::: "memory");
    }
    __syncthreads();

    // ---- fragments per ISA 7.12.2 layouts ----
    FragU af[2], bfr[4];
#pragma unroll
    for (int i = 0; i < 2; ++i) {
      const unsigned short* pa = &sA[cur][(wm * 32 + i * 16 + li) * LDS_K];
      af[i].p.lo = *(const v8bf*)(pa + g * 8);        // K = g*8 .. g*8+7
      af[i].p.hi = *(const v8bf*)(pa + 16 + g * 8);   // K = 16+g*8 ..
    }
#pragma unroll
    for (int j = 0; j < 4; ++j) {
      const unsigned short* pb = &sB[cur][(wn * 64 + j * 16 + li) * LDS_K + g * 8];
      bfr[j].p.lo = *(const v8bf*)(pb);               // K = g*8 .. g*8+7
      bfr[j].p.hi = *(const v8bf*)(pb + 16);          // K = 16+g*8 ..
    }

#pragma unroll
    for (int i = 0; i < 2; ++i)
#pragma unroll
      for (int j = 0; j < 4; ++j)
        acc[i][j] = __builtin_amdgcn_wmma_f32_16x16x32_bf16(
            false, af[i].v, false, bfr[j].v, (short)0, acc[i][j], false, false);

    __syncthreads();   // all waves done reading buffer `cur` before it is re-filled
  }

  // ---- store: D layout (lane group g -> M+8; VGPR r -> M row) ----
#pragma unroll
  for (int i = 0; i < 2; ++i) {
    const int mbase = row0 + wm * 32 + i * 16 + 8 * g;
#pragma unroll
    for (int j = 0; j < 4; ++j) {
      const int ncol = col0 + wn * 64 + j * 16 + li;
      if (ncol < N) {
#pragma unroll
        for (int r = 0; r < 8; ++r)
          C[(size_t)(mbase + r) * N + ncol] = acc[i][j][r];
        if (Cbf) {
#pragma unroll
          for (int r = 0; r < 8; ++r)
            Cbf[(size_t)(mbase + r) * N + ncol] = f32_to_bf16(acc[i][j][r]);
        }
      }
    }
  }
}

// ---------------- pack fp32 -> bf16 (same layout) ----------------
__global__ __launch_bounds__(256) void pack_bf16_kernel(
    const float* __restrict__ in, unsigned short* __restrict__ out, size_t count)
{
  const size_t i = (size_t)blockIdx.x * blockDim.x + threadIdx.x;
  if (i < count) out[i] = f32_to_bf16(in[i]);
}

// ---------------- pack + transpose: fp32 [K][N] -> bf16 [N][K] ----------------
__global__ __launch_bounds__(256) void pack_transpose_bf16_kernel(
    const float* __restrict__ in, unsigned short* __restrict__ out, int K, int N)
{
  const size_t i = (size_t)blockIdx.x * blockDim.x + threadIdx.x;
  if (i >= (size_t)K * N) return;
  const int n = (int)(i / K);
  const int k = (int)(i % K);
  out[i] = f32_to_bf16(in[(size_t)k * N + n]);
}

// ---------------- depthwise causal conv4 + bias + SiLU (fp32 + bf16 out) ----------------
__global__ __launch_bounds__(256) void conv_silu_kernel(
    const float* __restrict__ xz, const float* __restrict__ conv_w,
    const float* __restrict__ conv_b, float* __restrict__ x_conv,
    unsigned short* __restrict__ x_conv_bf)
{
  const size_t idx = (size_t)blockIdx.x * blockDim.x + threadIdx.x;
  if (idx >= (size_t)BATCH * SEQ * D_INNER) return;
  const int d = (int)(idx % D_INNER);
  const int s = (int)((idx / D_INNER) % SEQ);
  const int b = (int)(idx / ((size_t)D_INNER * SEQ));

  const float* base = xz + (size_t)b * SEQ * (2 * D_INNER);
  float acc = conv_b[d];
#pragma unroll
  for (int i = 0; i < D_CONV; ++i) {
    const int ss = s - (D_CONV - 1) + i;
    if (ss >= 0) acc = fmaf(conv_w[d * D_CONV + i], base[(size_t)ss * (2 * D_INNER) + d], acc);
  }
  const float sg = 1.0f / (1.0f + __expf(-acc));
  const float r = acc * sg;
  x_conv[idx] = r;
  x_conv_bf[idx] = f32_to_bf16(r);
}

// ---------------- dt activation ----------------
__global__ __launch_bounds__(256) void dt_act_kernel(
    float* __restrict__ dt, const float* __restrict__ b_dt)
{
  const size_t idx = (size_t)blockIdx.x * blockDim.x + threadIdx.x;
  if (idx >= (size_t)BATCH * SEQ * D_INNER) return;
  const int d = (int)(idx % D_INNER);
  const float v = dt[idx] + b_dt[d];
  const float s = 1.0f / (1.0f + __expf(-v));
  float r = s * (0.1f - 0.001f) + 0.001f;
  dt[idx] = fmaxf(r, 0.0001f);
}

// ---------------- selective scan over SEQ ----------------
__global__ __launch_bounds__(256) void scan_kernel(
    const float* __restrict__ dt, const float* __restrict__ xdbl,
    const float* __restrict__ xconv, const float* __restrict__ A_log,
    float* __restrict__ y)
{
  const int bpb = D_INNER / 256;                   // 8
  const int b = blockIdx.x / bpb;
  const int d = (blockIdx.x % bpb) * 256 + threadIdx.x;

  __shared__ float sBC[2 * D_STATE];

  float Arow[D_STATE];
#pragma unroll
  for (int n = 0; n < D_STATE; ++n)
    Arow[n] = -__expf(A_log[(size_t)d * D_STATE + n]);

  float h[D_STATE];
#pragma unroll
  for (int n = 0; n < D_STATE; ++n) h[n] = 0.0f;

  for (int t = 0; t < SEQ; ++t) {
    if (threadIdx.x < 2 * D_STATE)
      sBC[threadIdx.x] = xdbl[((size_t)b * SEQ + t) * XDBL_W + DT_RANK + threadIdx.x];
    __syncthreads();

    const size_t off = ((size_t)b * SEQ + t) * D_INNER + d;
    const float dtv = dt[off];
    const float du  = dtv * xconv[off];
    float yv = 0.0f;
#pragma unroll
    for (int n = 0; n < D_STATE; ++n) {
      h[n] = fmaf(__expf(dtv * Arow[n]), h[n], du * sBC[n]);
      yv = fmaf(h[n], sBC[D_STATE + n], yv);
    }
    y[off] = yv;
    __syncthreads();
  }
}

// ---------------- epilogue: yb16 = bf16((y + x_conv*D) * silu(z)) ----------------
__global__ __launch_bounds__(256) void epilogue_kernel(
    const float* __restrict__ y, const float* __restrict__ xconv,
    const float* __restrict__ xz, const float* __restrict__ D_param,
    unsigned short* __restrict__ ybf)
{
  const size_t idx = (size_t)blockIdx.x * blockDim.x + threadIdx.x;
  if (idx >= (size_t)BATCH * SEQ * D_INNER) return;
  const int d = (int)(idx % D_INNER);
  const size_t row = idx / D_INNER;                // b*SEQ + s
  const float z = xz[row * (2 * D_INNER) + D_INNER + d];
  const float sz = z / (1.0f + __expf(-z));
  ybf[idx] = f32_to_bf16(fmaf(xconv[idx], D_param[d], y[idx]) * sz);
}

// ---------------- launcher ----------------
extern "C" void kernel_launch(void* const* d_in, const int* in_sizes, int n_in,
                              void* d_out, int out_size, void* d_ws, size_t ws_size,
                              hipStream_t stream) {
  (void)in_sizes; (void)n_in; (void)out_size; (void)ws_size;
  const float* x       = (const float*)d_in[0];
  const float* W_in    = (const float*)d_in[1];
  const float* conv_w  = (const float*)d_in[2];
  const float* conv_b  = (const float*)d_in[3];
  const float* W_x     = (const float*)d_in[4];
  const float* W_dt    = (const float*)d_in[5];
  const float* b_dt    = (const float*)d_in[6];
  const float* A_log   = (const float*)d_in[7];
  const float* D_param = (const float*)d_in[8];
  const float* W_out   = (const float*)d_in[9];
  float* out = (float*)d_out;

  // ---- fp32 workspace ----
  float* ws    = (float*)d_ws;
  float* xz    = ws;                                            // 33,554,432
  float* xconv = xz    + (size_t)BATCH * SEQ * 2 * D_INNER;     // 16,777,216
  float* xdbl  = xconv + (size_t)BATCH * SEQ * D_INNER;         //    786,432
  float* dt    = xdbl  + (size_t)BATCH * SEQ * XDBL_W;          // 16,777,216
  float* ybuf  = dt    + (size_t)BATCH * SEQ * D_INNER;         // 16,777,216
  // ---- bf16 workspace (16B-aligned: all sub-sizes are multiples of 8) ----
  unsigned short* u16 = (unsigned short*)(ybuf + (size_t)BATCH * SEQ * D_INNER);
  unsigned short* xbf     = u16;                                      // 8,388,608
  unsigned short* WinT    = xbf    + (size_t)MROWS * D_MODEL;         // 4,194,304
  unsigned short* WxT     = WinT   + (size_t)(2 * D_INNER) * D_MODEL; //   196,608
  unsigned short* WdtT    = WxT    + (size_t)XDBL_W * D_INNER;        //   131,072
  unsigned short* WoutT   = WdtT   + (size_t)D_INNER * DT_RANK;       // 2,097,152
  unsigned short* xconvBf = WoutT  + (size_t)D_MODEL * D_INNER;       // 16,777,216
  unsigned short* xdblBf  = xconvBf + (size_t)MROWS * D_INNER;        //    786,432
  unsigned short* ybf     = xdblBf  + (size_t)MROWS * XDBL_W;         // 16,777,216

  const int EB = (BATCH * SEQ * D_INNER) / 256;                 // 65536 blocks

  // ---- 0) pack operands to bf16 (weights transposed to [N][K]) ----
  pack_bf16_kernel<<<(MROWS * D_MODEL) / 256, 256, 0, stream>>>(x, xbf, (size_t)MROWS * D_MODEL);
  pack_transpose_bf16_kernel<<<(D_MODEL * 2 * D_INNER) / 256, 256, 0, stream>>>(W_in, WinT, D_MODEL, 2 * D_INNER);
  pack_transpose_bf16_kernel<<<(D_INNER * XDBL_W) / 256, 256, 0, stream>>>(W_x, WxT, D_INNER, XDBL_W);
  pack_transpose_bf16_kernel<<<(DT_RANK * D_INNER) / 256, 256, 0, stream>>>(W_dt, WdtT, DT_RANK, D_INNER);
  pack_transpose_bf16_kernel<<<(D_INNER * D_MODEL) / 256, 256, 0, stream>>>(W_out, WoutT, D_INNER, D_MODEL);

  // ---- 1) xz = x @ W_in                M=8192 K=1024 N=4096 ----
  {
    dim3 grid((2 * D_INNER) / BN, MROWS / BM);
    wmma_gemm_bf16_kernel<<<grid, 256, 0, stream>>>(xbf, WinT, xz, nullptr,
                                                    MROWS, 2 * D_INNER, D_MODEL, D_MODEL);
  }
  // ---- 2) depthwise conv + SiLU (fp32 + bf16) ----
  conv_silu_kernel<<<EB, 256, 0, stream>>>(xz, conv_w, conv_b, xconv, xconvBf);
  // ---- 3) x_dbl = x_conv @ W_x         M=8192 K=2048 N=96 (+bf16 copy) ----
  {
    dim3 grid((XDBL_W + BN - 1) / BN, MROWS / BM);
    wmma_gemm_bf16_kernel<<<grid, 256, 0, stream>>>(xconvBf, WxT, xdbl, xdblBf,
                                                    MROWS, XDBL_W, D_INNER, D_INNER);
  }
  // ---- 4) dt_raw = dt_low @ W_dt       M=8192 K=64 N=2048 (A = strided bf16 slice) ----
  {
    dim3 grid(D_INNER / BN, MROWS / BM);
    wmma_gemm_bf16_kernel<<<grid, 256, 0, stream>>>(xdblBf, WdtT, dt, nullptr,
                                                    MROWS, D_INNER, DT_RANK, XDBL_W);
  }
  // ---- 5) dt activation (in place) ----
  dt_act_kernel<<<EB, 256, 0, stream>>>(dt, b_dt);
  // ---- 6) selective scan ----
  scan_kernel<<<BATCH * (D_INNER / 256), 256, 0, stream>>>(dt, xdbl, xconv, A_log, ybuf);
  // ---- 7) gated epilogue -> bf16 ----
  epilogue_kernel<<<EB, 256, 0, stream>>>(ybuf, xconv, xz, D_param, ybf);
  // ---- 8) out = y @ W_out              M=8192 K=2048 N=1024 ----
  {
    dim3 grid(D_MODEL / BN, MROWS / BM);
    wmma_gemm_bf16_kernel<<<grid, 256, 0, stream>>>(ybf, WoutT, out, nullptr,
                                                    MROWS, D_MODEL, D_INNER, D_INNER);
  }
}
